// _ModulatedConv2d_21088289423924
// MI455X (gfx1250) — matrix operand order repaired and checked by reading, not compiled, not run on hardware
//
#include <hip/hip_runtime.h>

typedef __attribute__((ext_vector_type(16))) __bf16 v16bf;
typedef __attribute__((ext_vector_type(8)))  float  v8f;
typedef int v4i __attribute__((vector_size(16)));

#define BATCH    8
#define IN_CH    512
#define OUT_CH   512
#define STYLEDIM 512
#define HH       64
#define WW       64

// conv_scale = 1/sqrt(512*9), lin_scale = 1/sqrt(512)
#define CONV_SCALE   0.0147313913f
#define CONV_SCALE2  2.1701389e-4f
#define LIN_SCALE    0.0441941738f

#define LSTR 40   // LDS row stride (ushort) for A tiles: 32 + 8 pad, 16B aligned
#define XSTR 40   // LDS row stride (ushort) for input halo tile

// CDNA5 async global->LDS copy path (ASYNCcnt-tracked), with sync fallback.
#if defined(__HIP_DEVICE_COMPILE__) && __has_builtin(__builtin_amdgcn_global_load_async_to_lds_b128)
#define HAVE_ASYNC_LDS 1
#else
#define HAVE_ASYNC_LDS 0
#endif

#if HAVE_ASYNC_LDS
typedef __attribute__((address_space(1))) v4i* gv4i_p;
typedef __attribute__((address_space(3))) v4i* lv4i_p;
#endif

__device__ __forceinline__ void cp16_to_lds(const unsigned short* src, unsigned short* dst) {
#if HAVE_ASYNC_LDS
  __builtin_amdgcn_global_load_async_to_lds_b128(
      (gv4i_p)(unsigned long long)src,
      (lv4i_p)(unsigned int)(unsigned long long)dst,
      0, 0);
#else
  *(uint4*)dst = *(const uint4*)src;
#endif
}

__device__ __forceinline__ void async_drain() {
#if HAVE_ASYNC_LDS
#if __has_builtin(__builtin_amdgcn_s_wait_asynccnt)
  __builtin_amdgcn_s_wait_asynccnt(0);
#else
  asm volatile("s_wait_asynccnt 0x0" ::: "memory");
#endif
#endif
}

// ---- float -> bf16 round-to-nearest-even ----
__device__ __forceinline__ unsigned short f2bf(float f) {
  union { float f; unsigned int u; } x; x.f = f;
  unsigned int u = x.u;
  if ((u & 0x7F800000u) == 0x7F800000u) return (unsigned short)(u >> 16); // inf/nan
  return (unsigned short)((u + 0x7FFFu + ((u >> 16) & 1u)) >> 16);
}

// ---- 1) modulation: s[b,i] = sum_j style[b,j]*mw[i,j]*lin_scale + mb[i] ----
__global__ __launch_bounds__(256) void mod_kernel(const float* __restrict__ style,
                                                  const float* __restrict__ mw,
                                                  const float* __restrict__ mb,
                                                  float* __restrict__ s_out) {
  const int b = blockIdx.x;
  __shared__ float st[STYLEDIM];
  for (int j = threadIdx.x; j < STYLEDIM; j += 256) st[j] = style[b * STYLEDIM + j];
  __syncthreads();
  for (int i = threadIdx.x; i < IN_CH; i += 256) {
    const float* row = mw + i * STYLEDIM;
    float acc = 0.f;
    for (int j = 0; j < STYLEDIM; ++j) acc = fmaf(st[j], row[j], acc);
    s_out[b * IN_CH + i] = acc * LIN_SCALE + mb[i];
  }
}

// ---- 2) demod[b,o] = rsqrt(cs2 * sum_i s[b,i]^2 * (sum_tap w[o,i,tap]^2) + 1e-8) ----
__global__ __launch_bounds__(256) void demod_kernel(const float* __restrict__ w,
                                                    const float* __restrict__ s,
                                                    float* __restrict__ demod) {
  const int o = blockIdx.x;
  const int t = threadIdx.x;
  float accb[BATCH];
#pragma unroll
  for (int b = 0; b < BATCH; ++b) accb[b] = 0.f;
  for (int i = t; i < IN_CH; i += 256) {
    const float* wp = w + (o * IN_CH + i) * 9;
    float wsq = 0.f;
#pragma unroll
    for (int k = 0; k < 9; ++k) { float v = wp[k]; wsq = fmaf(v, v, wsq); }
#pragma unroll
    for (int b = 0; b < BATCH; ++b) {
      float sv = s[b * IN_CH + i];
      accb[b] = fmaf(sv * sv, wsq, accb[b]);
    }
  }
  __shared__ float red[256 * BATCH];
#pragma unroll
  for (int b = 0; b < BATCH; ++b) red[t * BATCH + b] = accb[b];
  __syncthreads();
  if (t < BATCH) {
    float sum = 0.f;
    for (int j = 0; j < 256; ++j) sum += red[j * BATCH + t];
    demod[t * OUT_CH + o] = rsqrtf(CONV_SCALE2 * sum + 1e-8f);
  }
}

// ---- 3) weight -> bf16, transposed [tap][o][ci], conv_scale folded in ----
__global__ __launch_bounds__(256) void wconv_kernel(const float* __restrict__ w,
                                                    unsigned short* __restrict__ wb) {
  int e = blockIdx.x * 256 + threadIdx.x;            // [0, 9*512*512)
  int tap = e / (OUT_CH * IN_CH);
  int rem = e - tap * OUT_CH * IN_CH;
  int o  = rem >> 9;
  int ci = rem & 511;
  wb[e] = f2bf(w[(o * IN_CH + ci) * 9 + tap] * CONV_SCALE);
}

// ---- 4) input pre-scale by s[b,ci] -> bf16, NCHW -> NHWC via LDS transpose ----
__global__ __launch_bounds__(256) void xscale_kernel(const float* __restrict__ in,
                                                     const float* __restrict__ s,
                                                     unsigned short* __restrict__ xb) {
  const int b   = blockIdx.z;
  const int y   = blockIdx.y;
  const int ci0 = blockIdx.x * 64;
  const int t   = threadIdx.x;
  __shared__ unsigned short tile[64 * 65];
#pragma unroll
  for (int g = 0; g < 16; ++g) {
    int e  = t + g * 256;         // 64 ci x 64 x
    int ci = e >> 6;
    int x  = e & 63;
    float v = in[((b * IN_CH + ci0 + ci) * HH + y) * WW + x] * s[b * IN_CH + ci0 + ci];
    tile[ci * 65 + x] = f2bf(v);
  }
  __syncthreads();
#pragma unroll
  for (int g = 0; g < 16; ++g) {
    int e  = t + g * 256;
    int x  = e >> 6;
    int ci = e & 63;
    xb[((b * HH + y) * WW + x) * IN_CH + ci0 + ci] = tile[ci * 65 + x];
  }
}

// ---- 5) implicit-GEMM conv via bf16 WMMA ----
// Block: 128 o-rows x 128 positions (2 image rows x 64 cols), 8 waves (4 m x 2 n).
// ci0 outer (input halo staged once, shared by all 9 taps); tap inner with
// double-buffered async A tiles. Each wave: 2x4 16x16 frags, 8 WMMAs per tap.
__global__ __launch_bounds__(256) void modconv_wmma(
    const unsigned short* __restrict__ wb,   // [9][512][512] bf16
    const unsigned short* __restrict__ xb,   // [8][64][64][512] bf16 (NHWC)
    const float* __restrict__ demod,         // [8][512]
    float* __restrict__ out)                 // [8][512][64][64]
{
  __shared__ __align__(16) unsigned short sA[2][128 * LSTR];   // ping-pong A tiles
  __shared__ __align__(16) unsigned short sX[(4 * 66) * XSTR]; // input halo: 4 rows x 66 cols x 32 ci

  const int t    = threadIdx.x;
  const int lane = t & 31;
  const int wid  = t >> 5;
  const int wm   = wid & 3;          // 32-row band of o
  const int wn   = wid >> 2;         // which of the 2 image rows
  const int b      = blockIdx.z;
  const int oBlock = blockIdx.y * 128;
  const int y0     = blockIdx.x * 2;

  v8f acc[2][4];
#pragma unroll
  for (int fm = 0; fm < 2; ++fm)
#pragma unroll
    for (int fn = 0; fn < 4; ++fn)
#pragma unroll
      for (int r = 0; r < 8; ++r) acc[fm][fn][r] = 0.f;

  const int half = lane >> 4;
  const int l16  = lane & 15;

  for (int ci0 = 0; ci0 < IN_CH; ci0 += 32) {
    __syncthreads();   // all reads of sX / sA[0] from previous ci0 complete

    // ---- stage input halo: 4 rows (y0-1..y0+2) x 66 cols x 32 ci, zero-padded ----
    for (int v = t; v < 1056; v += 256) {      // 264 positions x 4 x 16B
      int pos = v >> 2, cv = v & 3;
      int yy  = pos / 66;
      int xx  = pos - yy * 66;
      int y   = y0 - 1 + yy;
      int x   = xx - 1;
      uint4 val = {0u, 0u, 0u, 0u};
      if ((unsigned)y < 64u && (unsigned)x < 64u)
        val = *(const uint4*)(xb + ((b * HH + y) * WW + x) * IN_CH + ci0 + cv * 8);
      *(uint4*)(&sX[pos * XSTR + cv * 8]) = val;
    }
    // ---- stage A tile for tap 0 (async) ----
#pragma unroll
    for (int g = 0; g < 2; ++g) {
      int v = t + g * 256;
      int row = v >> 2, cv = v & 3;
      cp16_to_lds(wb + (oBlock + row) * IN_CH + ci0 + cv * 8,
                  &sA[0][row * LSTR + cv * 8]);
    }
    async_drain();
    __syncthreads();

    for (int tap = 0; tap < 9; ++tap) {
      const int p  = tap & 1;
      const int kh = tap / 3 - 1;
      const int kw = tap % 3 - 1;

      // prefetch next tap's A tile into the other buffer (overlaps the WMMAs)
      if (tap < 8) {
        const unsigned short* wnext = wb + (tap + 1) * (OUT_CH * IN_CH);
#pragma unroll
        for (int g = 0; g < 2; ++g) {
          int v = t + g * 256;
          int row = v >> 2, cv = v & 3;
          cp16_to_lds(wnext + (oBlock + row) * IN_CH + ci0 + cv * 8,
                      &sA[1 - p][row * LSTR + cv * 8]);
        }
      }

      // A fragments: 16-bit A 16x32 layout — lanes 0-15: K0..7 + K16..23,
      // lanes 16-31: K8..15 + K24..31 (8-element chunks)
      v16bf af[2];
#pragma unroll
      for (int fm = 0; fm < 2; ++fm) {
        int r = wm * 32 + fm * 16 + l16;
        union { uint4 u[2]; v16bf v; } fu;
        fu.u[0] = *(const uint4*)(&sA[p][r * LSTR + half * 8]);
        fu.u[1] = *(const uint4*)(&sA[p][r * LSTR + 16 + half * 8]);
        af[fm] = fu.v;
      }
      // B fragments from halo: lanes 0-15 hold K0..7 + K16..23, lanes 16-31 hold K8..15 + K24..31
      const int yyb = (wn + kh + 1) * 66;      // halo row base for this wave
      v16bf bfr[4];
#pragma unroll
      for (int fn = 0; fn < 4; ++fn) {
        int pos = yyb + fn * 16 + l16 + kw + 1;
        union { uint4 u[2]; v16bf v; } fu;
        fu.u[0] = *(const uint4*)(&sX[pos * XSTR + half * 8]);
        fu.u[1] = *(const uint4*)(&sX[pos * XSTR + 16 + half * 8]);
        bfr[fn] = fu.v;
      }

#pragma unroll
      for (int fm = 0; fm < 2; ++fm)
#pragma unroll
        for (int fn = 0; fn < 4; ++fn)
          acc[fm][fn] = __builtin_amdgcn_wmma_f32_16x16x32_bf16(
              false, af[fm], false, bfr[fn], (short)0, acc[fm][fn], false, false);

      if (tap < 8) async_drain();   // next-tap A tile landed in LDS
      __syncthreads();
    }
  }

  // epilogue: D layout — M = r + 8*(lane>=16), N = lane&15 ; scale by demod[b,o]
#pragma unroll
  for (int fm = 0; fm < 2; ++fm) {
#pragma unroll
    for (int r = 0; r < 8; ++r) {
      int o = oBlock + wm * 32 + fm * 16 + r + half * 8;
      float dm = demod[b * OUT_CH + o];
      int y = y0 + wn;
#pragma unroll
      for (int fn = 0; fn < 4; ++fn) {
        int x = fn * 16 + l16;
        out[((b * OUT_CH + o) * HH + y) * WW + x] = acc[fm][fn][r] * dm;
      }
    }
  }
}

extern "C" void kernel_launch(void* const* d_in, const int* in_sizes, int n_in,
                              void* d_out, int out_size, void* d_ws, size_t ws_size,
                              hipStream_t stream) {
  const float* input  = (const float*)d_in[0];
  const float* style  = (const float*)d_in[1];
  const float* weight = (const float*)d_in[2];
  const float* mod_w  = (const float*)d_in[3];
  const float* mod_b  = (const float*)d_in[4];
  float* out = (float*)d_out;

  // workspace layout (~36.5 MB): s | demod | wb(bf16) | xb(bf16 NHWC)
  float* s     = (float*)d_ws;
  float* demod = s + BATCH * IN_CH;
  unsigned short* wb = (unsigned short*)(demod + BATCH * OUT_CH);
  unsigned short* xb = wb + 9 * OUT_CH * IN_CH;

  mod_kernel  <<<BATCH, 256, 0, stream>>>(style, mod_w, mod_b, s);
  demod_kernel<<<OUT_CH, 256, 0, stream>>>(weight, s, demod);
  wconv_kernel<<<(9 * OUT_CH * IN_CH) / 256, 256, 0, stream>>>(weight, wb);
  xscale_kernel<<<dim3(IN_CH / 64, HH, BATCH), 256, 0, stream>>>(input, s, xb);

  dim3 grid(32, 4, BATCH);  // n-tiles x m-tiles x batch = 1024 blocks
  modconv_wmma<<<grid, 256, 0, stream>>>(wb, xb, demod, out);
}
